// SupernodePooling_47931835023950
// MI455X (gfx1250) — compile-verified, matrix operand order checked by reading, not run on hardware
//
#include <hip/hip_runtime.h>
#include <math.h>
#include <stdint.h>

typedef _Float16 half_t;
typedef _Float16 v16h __attribute__((ext_vector_type(16)));
typedef _Float16 v8h  __attribute__((ext_vector_type(8)));
typedef float    v8f  __attribute__((ext_vector_type(8)));

#define HDIM   192
#define IN_DIM 16
#define NPTS   262144
#define SSEG   16384
#define DEG    32
#define KDIM1  384          // 2*H
#define KT1    12           // 384/32 k-steps for matmul1
#define KT2    6            // 192/32 k-steps for matmul2
#define NTILES 12           // 192/16 n-tiles
#define APITCH 392          // padded LDS row pitch (halves): 196 dwords -> conflict-free
#define SB     (32 * APITCH) // per-supernode LDS A block (halves)

// ---------------------------------------------------------------------------
// Pack a row-major f32 weight matrix [K, 192] into f16 WMMA B-fragments.
// Fragment (kt,nt), lane l, slot h -> k = kt*32+(h/8)*16+(l/16)*8+(h%8),
//                                     n = nt*16 + l%16
// ---------------------------------------------------------------------------
__global__ void pack_w(const float* __restrict__ W, half_t* __restrict__ Wp, int total) {
    int idx = blockIdx.x * 256 + threadIdx.x;
    if (idx >= total) return;
    int f  = idx >> 9;          // fragment id = kt*NTILES + nt
    int l  = (idx >> 4) & 31;   // lane
    int h  = idx & 15;          // half slot
    int kt = f / NTILES, nt = f % NTILES;
    int n  = nt * 16 + (l & 15);
    int k  = kt * 32 + (h >> 3) * 16 + (l >> 4) * 8 + (h & 7);
    Wp[idx] = (half_t)W[k * HDIM + n];
}

// ---------------------------------------------------------------------------
// Kernel 1: x = feat @ W_in + b_in + sincos_embed(pos), stored f16 [N, 192].
// ---------------------------------------------------------------------------
__global__ void node_embed(const float* __restrict__ feat, const float* __restrict__ pos,
                           const float* __restrict__ W_in, const float* __restrict__ b_in,
                           half_t* __restrict__ x) {
    __shared__ float sf[IN_DIM];
    __shared__ float sp[3];
    int nidx = blockIdx.x;
    int h    = threadIdx.x;                 // 0..191
    if (h < IN_DIM) sf[h] = feat[nidx * IN_DIM + h];
    if (h < 3)      sp[h] = pos[nidx * 3 + h];
    __syncthreads();
    float acc = b_in[h];
#pragma unroll
    for (int i = 0; i < IN_DIM; ++i) acc += sf[i] * W_in[i * HDIM + h];
    int d = h / 64, j = h % 64;
    int kk = (j < 32) ? j : (j - 32);
    float omega = exp2f(-(2.0f * (float)kk / 64.0f) * 13.287712379549449f); // log2(10000)
    float t = sp[d] * omega;
    acc += (j < 32) ? sinf(t) : cosf(t);
    x[(size_t)nidx * HDIM + h] = (half_t)acc;
}

// ---------------------------------------------------------------------------
// Kernel 2: 2 supernodes per workgroup (128 threads, 4 waves).
//   A = concat(x[src_e], x[dst]) for 2*32 edges -> LDS via async-to-LDS DMA
//   h1 = GELU(A @ W1 + b1) via v_wmma_f32_16x16x32_f16, B fragment shared
//        across 4 WMMAs (2 supernodes x 2 M-tiles) -> halves W1 L2 traffic
//   hbar[s] = mean_rows(h1) reduced in-register
// ---------------------------------------------------------------------------
__launch_bounds__(128)
__global__ void edge_mlp_pool(const half_t* __restrict__ x,
                              const int* __restrict__ src_idx,
                              const int* __restrict__ sup_idx,
                              const half_t* __restrict__ W1p,
                              const float* __restrict__ b1,
                              half_t* __restrict__ hbar) {
    __shared__ __align__(16) half_t A[2 * SB];
    int s0  = blockIdx.x * 2;
    int tid = threadIdx.x;

    // Stage A for both supernodes: 2 x 32 rows x 384 halves in 16B chunks.
    // 3072 chunks / 128 threads = 24 async-to-LDS b128 ops per lane.
    for (int c = tid; c < 3072; c += 128) {
        int ss   = c / 1536;              // local supernode 0/1
        int cc0  = c % 1536;
        int row  = cc0 / 48;
        int cc   = cc0 % 48;
        int part = cc / 24;               // 0 = src half, 1 = dst half
        int j    = cc % 24;               // 8-half chunk within 192-col half
        int s    = s0 + ss;
        int node = part ? sup_idx[s] : src_idx[s * DEG + row];
        const half_t* gp = x + (size_t)node * HDIM + j * 8;
        // generic LDS pointer truncates to the 32-bit LDS byte offset
        uint32_t laddr = (uint32_t)(uintptr_t)(A + ss * SB + row * APITCH + part * HDIM + j * 8);
        asm volatile("global_load_async_to_lds_b128 %0, %1, off"
                     :: "v"(laddr), "v"((uint64_t)(uintptr_t)gp)
                     : "memory");
    }
    asm volatile("s_wait_asynccnt 0x0" ::: "memory");
    __syncthreads();

    int wave = tid >> 5;
    int lane = tid & 31;
    int ln16 = lane & 15;
    int lhi  = lane >> 4;

#pragma unroll
    for (int t = 0; t < 3; ++t) {
        int nt = wave * 3 + t;
        v8f a00 = {0.f,0.f,0.f,0.f,0.f,0.f,0.f,0.f};  // s0, rows 0..15
        v8f a01 = {0.f,0.f,0.f,0.f,0.f,0.f,0.f,0.f};  // s0, rows 16..31
        v8f a10 = {0.f,0.f,0.f,0.f,0.f,0.f,0.f,0.f};  // s1, rows 0..15
        v8f a11 = {0.f,0.f,0.f,0.f,0.f,0.f,0.f,0.f};  // s1, rows 16..31
        for (int kt = 0; kt < KT1; ++kt) {
            v16h b = *(const v16h*)(W1p + ((size_t)(kt * NTILES + nt) * 32 + lane) * 16);
            const half_t* base = A + ln16 * APITCH + kt * 32 + lhi * 8;
#pragma unroll
            for (int q = 0; q < 4; ++q) {   // q = ss*2 + mt
                const half_t* ap = base + q * (16 * APITCH);
                v8h lo = *(const v8h*)ap;
                v8h hi = *(const v8h*)(ap + 16);
                v16h af = __builtin_shufflevector(lo, hi, 0,1,2,3,4,5,6,7,8,9,10,11,12,13,14,15);
                switch (q) {
                case 0: a00 = __builtin_amdgcn_wmma_f32_16x16x32_f16(false, af, false, b, (short)0, a00, false, false); break;
                case 1: a01 = __builtin_amdgcn_wmma_f32_16x16x32_f16(false, af, false, b, (short)0, a01, false, false); break;
                case 2: a10 = __builtin_amdgcn_wmma_f32_16x16x32_f16(false, af, false, b, (short)0, a10, false, false); break;
                case 3: a11 = __builtin_amdgcn_wmma_f32_16x16x32_f16(false, af, false, b, (short)0, a11, false, false); break;
                }
            }
        }
        // bias + exact GELU + row mean.  D layout: lane = n + 16*(m/8), reg = m%8.
        float bn   = b1[nt * 16 + ln16];
        float sum0 = 0.f, sum1 = 0.f;
#pragma unroll
        for (int r = 0; r < 8; ++r) {
            float v0 = a00[r] + bn;
            float v1 = a01[r] + bn;
            float v2 = a10[r] + bn;
            float v3 = a11[r] + bn;
            sum0 += 0.5f * v0 * (1.0f + erff(v0 * 0.70710678118654752f));
            sum0 += 0.5f * v1 * (1.0f + erff(v1 * 0.70710678118654752f));
            sum1 += 0.5f * v2 * (1.0f + erff(v2 * 0.70710678118654752f));
            sum1 += 0.5f * v3 * (1.0f + erff(v3 * 0.70710678118654752f));
        }
        sum0 += __shfl_xor(sum0, 16, 32);   // combine row-halves (lanes L / L+16)
        sum1 += __shfl_xor(sum1, 16, 32);
        if (lhi == 0) {
            hbar[(size_t)s0 * HDIM + nt * 16 + ln16]       = (half_t)(sum0 * (1.0f / 32.0f));
            hbar[(size_t)(s0 + 1) * HDIM + nt * 16 + ln16] = (half_t)(sum1 * (1.0f / 32.0f));
        }
    }
}

// ---------------------------------------------------------------------------
// Kernel 3: pooled = hbar @ W2 + b2  (f16 in, f32 out). [16384,192]x[192,192]
// ---------------------------------------------------------------------------
__launch_bounds__(128)
__global__ void final_gemm(const half_t* __restrict__ hbar,
                           const half_t* __restrict__ W2p,
                           const float* __restrict__ b2,
                           float* __restrict__ out) {
    int mt   = blockIdx.x;                 // 0..1023
    int tid  = threadIdx.x;
    int wave = tid >> 5;
    int lane = tid & 31;
    int ln16 = lane & 15;
    int lhi  = lane >> 4;
    const half_t* arow = hbar + (size_t)(mt * 16 + ln16) * HDIM + lhi * 8;

#pragma unroll
    for (int t = 0; t < 3; ++t) {
        int nt = wave * 3 + t;
        v8f acc = {0.f,0.f,0.f,0.f,0.f,0.f,0.f,0.f};
        for (int kt = 0; kt < KT2; ++kt) {
            const half_t* ap = arow + kt * 32;
            v8h alo = *(const v8h*)ap;
            v8h ahi = *(const v8h*)(ap + 16);
            v16h a = __builtin_shufflevector(alo, ahi, 0,1,2,3,4,5,6,7,8,9,10,11,12,13,14,15);
            v16h b = *(const v16h*)(W2p + ((size_t)(kt * NTILES + nt) * 32 + lane) * 16);
            acc = __builtin_amdgcn_wmma_f32_16x16x32_f16(false, a, false, b, (short)0, acc, false, false);
        }
        float bn  = b2[nt * 16 + ln16];
        int mbase = mt * 16 + lhi * 8;
#pragma unroll
        for (int r = 0; r < 8; ++r)
            out[(size_t)(mbase + r) * HDIM + nt * 16 + ln16] = acc[r] + bn;
    }
}

// ---------------------------------------------------------------------------
extern "C" void kernel_launch(void* const* d_in, const int* in_sizes, int n_in,
                              void* d_out, int out_size, void* d_ws, size_t ws_size,
                              hipStream_t stream) {
    const float* input_feat = (const float*)d_in[0];
    const float* input_pos  = (const float*)d_in[1];
    const int*   sup_idx    = (const int*)d_in[2];
    /* batch_idx d_in[3] unused */
    const int*   src_idx    = (const int*)d_in[4];
    /* dst_idx  d_in[5] unused: dst = repeat(supernode_idxs, 32) */
    const float* W_in = (const float*)d_in[6];
    const float* b_in = (const float*)d_in[7];
    const float* W1   = (const float*)d_in[8];
    const float* b1   = (const float*)d_in[9];
    const float* W2   = (const float*)d_in[10];
    const float* b2   = (const float*)d_in[11];
    float* out = (float*)d_out;

    char* ws = (char*)d_ws;
    half_t* x    = (half_t*)ws;                                       // N*H f16   (~96 MB)
    half_t* hbar = (half_t*)(ws + (size_t)NPTS * HDIM * 2);           // S*H f16   (~6 MB)
    half_t* W1p  = (half_t*)(ws + (size_t)(NPTS + SSEG) * HDIM * 2);  // 384*192 f16
    half_t* W2p  = (half_t*)((char*)W1p + (size_t)KDIM1 * HDIM * 2);  // 192*192 f16

    pack_w<<<(KT1 * NTILES * 512 + 255) / 256, 256, 0, stream>>>(W1, W1p, KT1 * NTILES * 512);
    pack_w<<<(KT2 * NTILES * 512 + 255) / 256, 256, 0, stream>>>(W2, W2p, KT2 * NTILES * 512);

    node_embed<<<NPTS, HDIM, 0, stream>>>(input_feat, input_pos, W_in, b_in, x);

    edge_mlp_pool<<<SSEG / 2, 128, 0, stream>>>(x, src_idx, sup_idx, W1p, b1, hbar);

    final_gemm<<<SSEG / 16, 128, 0, stream>>>(hbar, W2p, b2, out);
}
